// RNN_Classifier2_85933705658665
// MI455X (gfx1250) — matrix operand order repaired and checked by reading, not compiled, not run on hardware
//
#include <hip/hip_runtime.h>
#include <hip/hip_bf16.h>

typedef __attribute__((ext_vector_type(16))) __bf16 v16bf;
typedef __attribute__((ext_vector_type(8)))  __bf16 v8bf;
typedef __attribute__((ext_vector_type(8)))  float  v8f;
typedef __attribute__((ext_vector_type(4)))  float  v4f;

#define HD      128     // hidden size
#define TSTEPS  784     // 28*28
#define MTILE   16      // batch rows per workgroup (WMMA M)
#define NTHREADS 256    // 8 wave32s

__device__ __forceinline__ float fast_rcp(float x) {
#if __has_builtin(__builtin_amdgcn_rcpf)
  return __builtin_amdgcn_rcpf(x);
#else
  return 1.0f / x;
#endif
}

__device__ __forceinline__ float fast_tanh(float x) {
#if __has_builtin(__builtin_amdgcn_tanhf)
  return __builtin_amdgcn_tanhf(x);
#else
  // tanh(x) = 1 - 2/(exp(2x)+1)
  return 1.0f - 2.0f * fast_rcp(__expf(2.0f * x) + 1.0f);
#endif
}

// sigmoid(x) = 0.5*tanh(x/2) + 0.5 : one TRANS op instead of exp+rcp
__device__ __forceinline__ float fast_sigmoid(float x) {
#if __has_builtin(__builtin_amdgcn_tanhf)
  return __builtin_fmaf(0.5f, __builtin_amdgcn_tanhf(0.5f * x), 0.5f);
#else
  return fast_rcp(1.0f + __expf(-x));
#endif
}

// One recurrent step: read h from hr, write new h into hw, single barrier.
// acc[g] for kc=0 consumes xg[g] directly as the WMMA C operand (D != C),
// so no per-step accumulator-init copies are generated.
__device__ __forceinline__ void lstm_step(
    const __bf16 (&hr)[MTILE][HD],
    __bf16 (&hw)[MTILE][HD],
    const v16bf (&bfrag)[4][4],
    const v8f (&xg)[4],
    float (&creg)[8],
    float (&hreg)[8],
    int l16, int half, int wave)
{
  v8f acc[4];

  // kc = 0 : acc = h_chunk0 @ W_chunk0 + xg
  {
    const int koff = half * 16;
    v8bf lo = *(const v8bf*)&hr[l16][koff];
    v8bf hi = *(const v8bf*)&hr[l16][koff + 8];
    v16bf a = __builtin_shufflevector(lo, hi,
               0,1,2,3,4,5,6,7, 8,9,10,11,12,13,14,15);
    #pragma unroll
    for (int g = 0; g < 4; ++g)
      acc[g] = __builtin_amdgcn_wmma_f32_16x16x32_bf16(
          false, a, false, bfrag[g][0], (short)0, xg[g], false, false);
  }

  // kc = 1..3 : chained accumulation
  #pragma unroll
  for (int kc = 1; kc < 4; ++kc) {
    const int koff = kc * 32 + half * 16;
    v8bf lo = *(const v8bf*)&hr[l16][koff];
    v8bf hi = *(const v8bf*)&hr[l16][koff + 8];
    v16bf a = __builtin_shufflevector(lo, hi,
               0,1,2,3,4,5,6,7, 8,9,10,11,12,13,14,15);
    #pragma unroll
    for (int g = 0; g < 4; ++g)
      acc[g] = __builtin_amdgcn_wmma_f32_16x16x32_bf16(
          false, a, false, bfrag[g][kc], (short)0, acc[g], false, false);
  }

  // LSTM cell update — fully in-lane (i/f/g/o share lane + VGPR index)
  #pragma unroll
  for (int r = 0; r < 8; ++r) {
    const float iv = fast_sigmoid(acc[0][r]);
    const float fv = fast_sigmoid(acc[1][r]);
    const float gv = fast_tanh   (acc[2][r]);
    const float ov = fast_sigmoid(acc[3][r]);
    const float cv = fv * creg[r] + iv * gv;
    creg[r] = cv;
    const float hv = ov * fast_tanh(cv);
    hreg[r] = hv;
    hw[r + 8 * half][wave * 16 + l16] = (__bf16)hv;
  }

  // Single barrier: step-t reads of hr retired before their consuming WMMAs,
  // so after this barrier hr is free to be overwritten next step, and hw is
  // visible to all waves.
  __syncthreads();
}

__global__ void __launch_bounds__(NTHREADS)
lstm_seq_kernel(const float* __restrict__ imgs,   // (B,28,28)
                const float* __restrict__ W_ih,   // (512,1)
                const float* __restrict__ W_hh,   // (512,128)
                const float* __restrict__ b_ih,   // (512)
                const float* __restrict__ b_hh,   // (512)
                const float* __restrict__ W_out,  // (10,128)
                const float* __restrict__ b_out,  // (10)
                float* __restrict__ out)          // (B,10)
{
  __shared__ __bf16 h_lds[2][MTILE][HD];  // 8 KB: double-buffered h exchange
  __shared__ float  hf_lds[MTILE][HD];    // 8 KB: final h (f32) for output GEMV

  const int tid  = threadIdx.x;
  const int wave = tid >> 5;        // 0..7 -> hidden slice [16*wave, 16*wave+16)
  const int lane = tid & 31;
  const int half = lane >> 4;       // 0/1: half-wave (selects K/M sub-block)
  const int l16  = lane & 15;
  const int m_base = blockIdx.x * MTILE;

  // ---------------------------------------------------------------
  // B fragments: W_hh^T tiles, f32->bf16, LIVE IN VGPRS for all steps.
  // B[k][n] = W_hh[n][k]; per-lane: column n = gate*128 + 16*wave + l16,
  // K = kc*32 + 16*half .. +15 (contiguous in W_hh row).
  // ---------------------------------------------------------------
  v16bf bfrag[4][4];                        // [gate][kchunk] : 128 VGPRs
  #pragma unroll
  for (int g = 0; g < 4; ++g) {
    const int n = g * HD + wave * 16 + l16;
    const float* wrow = W_hh + (size_t)n * HD;
    #pragma unroll
    for (int kc = 0; kc < 4; ++kc) {
      const int k0 = kc * 32 + half * 16;
      const v4f* wp = (const v4f*)(wrow + k0);
      v16bf f;
      #pragma unroll
      for (int q = 0; q < 4; ++q) {
        v4f tv = wp[q];
        f[4*q+0] = (__bf16)tv[0];
        f[4*q+1] = (__bf16)tv[1];
        f[4*q+2] = (__bf16)tv[2];
        f[4*q+3] = (__bf16)tv[3];
      }
      bfrag[g][kc] = f;
    }
  }

  // ---------------------------------------------------------------
  // xg fragments (C/D layout: lane -> N=l16, VGPR r -> M=r+8*half).
  // xg[b,n] = x0[b]*W_ih[n] + b_ih[n] + b_hh[n], constant over t.
  // ---------------------------------------------------------------
  float x0v[8];
  #pragma unroll
  for (int r = 0; r < 8; ++r)
    x0v[r] = imgs[(size_t)(m_base + r + 8 * half) * TSTEPS];   // first pixel

  v8f xg[4];
  #pragma unroll
  for (int g = 0; g < 4; ++g) {
    const int n = g * HD + wave * 16 + l16;
    const float wih  = W_ih[n];
    const float bias = b_ih[n] + b_hh[n];
    #pragma unroll
    for (int r = 0; r < 8; ++r) xg[g][r] = x0v[r] * wih + bias;
  }

  // persistent cell state (C/D layout slice); h=0 in read buffer 0
  float creg[8], hreg[8];
  #pragma unroll
  for (int r = 0; r < 8; ++r) { creg[r] = 0.0f; hreg[r] = 0.0f; }
  for (int idx = tid; idx < MTILE * HD; idx += NTHREADS)
    ((__bf16*)h_lds[0])[idx] = (__bf16)0.0f;
  __syncthreads();

  // ---------------------------------------------------------------
  // 784 recurrent steps, two per iteration (ping-pong buffers)
  // ---------------------------------------------------------------
  #pragma unroll 1
  for (int t = 0; t < TSTEPS; t += 2) {
    lstm_step(h_lds[0], h_lds[1], bfrag, xg, creg, hreg, l16, half, wave);
    lstm_step(h_lds[1], h_lds[0], bfrag, xg, creg, hreg, l16, half, wave);
  }

  // ---------------------------------------------------------------
  // out = h @ W_out^T + b_out   (tiny GEMV, one-time)
  // ---------------------------------------------------------------
  #pragma unroll
  for (int r = 0; r < 8; ++r)
    hf_lds[r + 8 * half][wave * 16 + l16] = hreg[r];
  __syncthreads();

  if (tid < MTILE * 10) {
    const int m = tid / 10, cls = tid % 10;
    const float* wrow = W_out + cls * HD;
    float sum = b_out[cls];
    #pragma unroll 4
    for (int k = 0; k < HD; ++k) sum += hf_lds[m][k] * wrow[k];
    out[(size_t)(m_base + m) * 10 + cls] = sum;
  }
}

extern "C" void kernel_launch(void* const* d_in, const int* in_sizes, int n_in,
                              void* d_out, int out_size, void* d_ws, size_t ws_size,
                              hipStream_t stream) {
  const float* imgs  = (const float*)d_in[0];
  const float* W_ih  = (const float*)d_in[1];
  const float* W_hh  = (const float*)d_in[2];
  const float* b_ih  = (const float*)d_in[3];
  const float* b_hh  = (const float*)d_in[4];
  const float* W_out = (const float*)d_in[5];
  const float* b_out = (const float*)d_in[6];

  const int B = in_sizes[0] / TSTEPS;          // 1024
  const int nblocks = B / MTILE;               // 64 workgroups, 8 waves each

  lstm_seq_kernel<<<dim3(nblocks), dim3(NTHREADS), 0, stream>>>(
      imgs, W_ih, W_hh, b_ih, b_hh, W_out, b_out, (float*)d_out);
}
